// MetabolicAttentionHead_74577812127868
// MI455X (gfx1250) — compile-verified, matrix operand order checked by reading, not run on hardware
//
#include <hip/hip_runtime.h>
#include <hip/hip_bf16.h>
#include <math.h>

typedef __attribute__((ext_vector_type(16))) _Float16 v16h;
typedef __attribute__((ext_vector_type(8)))  float    v8f;

#define D_MODEL 1024
#define HEAD_DIM 64
#define NB 4
#define T 4096
#define ROWS (NB * T)                 // 16384
#define OUT_OFF   0
#define ATTN_OFF  (ROWS * HEAD_DIM)   // 1048576
#define EN_OFF    (ATTN_OFF + (size_t)ROWS * T) // 68157440

// ---- WMMA fragment helpers (CDNA5 16x16x32 f16 layouts, ISA 7.12.2) ----

__device__ __forceinline__ int frag_koff(int j, int half) {
    // VGPR j (packing f16 elements 2j,2j+1) holds K = base + {0,1}
    return ((j & 4) ? 16 : 0) + half * 8 + (j & 3) * 2;
}

// A-frag (16xK row-major source) OR B-frag when B[k][n] == Src[n][k] (row gather).
__device__ __forceinline__ v16h frag_row_f32(const float* __restrict__ rowp, int kb, int half) {
    v16h f;
#pragma unroll
    for (int j = 0; j < 8; ++j) {
        int k = kb + frag_koff(j, half);
        f[2 * j]     = (_Float16)rowp[k];
        f[2 * j + 1] = (_Float16)rowp[k + 1];
    }
    return f;
}

__device__ __forceinline__ v16h frag_row_f16(const _Float16* rowp, int kb, int half) {
    v16h f;
#pragma unroll
    for (int j = 0; j < 8; ++j) {
        int k = kb + frag_koff(j, half);
        f[2 * j]     = rowp[k];
        f[2 * j + 1] = rowp[k + 1];
    }
    return f;
}

// B-frag when source is row-major [K, N] (lane's column fixed, walk down K).
__device__ __forceinline__ v16h frag_colB_f16(const _Float16* __restrict__ base, int ld,
                                              int col, int kb, int half) {
    v16h f;
#pragma unroll
    for (int j = 0; j < 8; ++j) {
        int k = kb + frag_koff(j, half);
        f[2 * j]     = base[(size_t)k * ld + col];
        f[2 * j + 1] = base[(size_t)(k + 1) * ld + col];
    }
    return f;
}

__device__ __forceinline__ v8f wmma_f16f32(v16h a, v16h b, v8f c) {
    return __builtin_amdgcn_wmma_f32_16x16x32_f16(false, a, false, b, (short)0, c, false, false);
}

// ---- Kernel 1: q/k/v = x @ W^T, stored as f16 (6 MB total -> stays in L2) ----
__global__ void __launch_bounds__(128)
proj_kernel(const float* __restrict__ x,
            const float* __restrict__ Wq, const float* __restrict__ Wk,
            const float* __restrict__ Wv,
            _Float16* __restrict__ qh, _Float16* __restrict__ kh, _Float16* __restrict__ vh) {
    int wid  = blockIdx.x * (blockDim.x >> 5) + (threadIdx.x >> 5);
    int lane = threadIdx.x & 31;
    int c = wid & 3;            // col tile (0..3) of 64
    int m = (wid >> 2) % 3;     // which projection
    int r = wid / 12;           // row tile (0..1023)
    const float* W  = (m == 0) ? Wq : (m == 1) ? Wk : Wv;
    _Float16*    Y  = (m == 0) ? qh : (m == 1) ? kh : vh;
    int half = lane >> 4;
    const float* xrow = x + (size_t)(r * 16 + (lane & 15)) * D_MODEL;
    const float* wrow = W + (size_t)(c * 16 + (lane & 15)) * D_MODEL; // B[k][n]=W[n][k]
    v8f acc = {};
    for (int kb = 0; kb < D_MODEL; kb += 32) {
        v16h a = frag_row_f32(xrow, kb, half);
        v16h b = frag_row_f32(wrow, kb, half);
        acc = wmma_f16f32(a, b, acc);
    }
    int n = lane & 15;
#pragma unroll
    for (int rr = 0; rr < 8; ++rr) {
        int row = r * 16 + rr + half * 8;
        Y[(size_t)row * HEAD_DIM + c * 16 + n] = (_Float16)acc[rr];
    }
}

// ---- Kernel 2: fused scores -> online softmax -> attn store -> PV -> entropy ----
// One wave owns 16 query rows. 4 batches * 256 qtiles = 1024 waves (8 waves/block).
// Pass 1: online per-lane row max + rescaled sum over 256 key tiles (2 WMMAs each).
// Pass 2: recompute, normalize, single attn write, LDS re-layout P (C-form -> A-form),
//         4 WMMAs vs V per key-tile pair, entropy accumulation.
__global__ void __launch_bounds__(256)
attn_fused_kernel(const _Float16* __restrict__ qh, const _Float16* __restrict__ kh,
                  const _Float16* __restrict__ vh,
                  float* __restrict__ attn, float* __restrict__ out,
                  float* __restrict__ ent_accum) {
    int wid  = blockIdx.x * 8 + (threadIdx.x >> 5);  // 0..1023
    int lane = threadIdx.x & 31;
    int qt = wid & 255;
    int b  = wid >> 8;
    int half = lane >> 4;
    int n = lane & 15;

    __shared__ _Float16 pbuf_all[8][16 * 32];        // per-wave 16x32 P staging (1 KB)
    _Float16* pbuf = pbuf_all[threadIdx.x >> 5];

    const _Float16* qrow  = qh + (size_t)(b * T + qt * 16 + n) * HEAD_DIM;
    const _Float16* kbase = kh + (size_t)b * T * HEAD_DIM;
    const _Float16* vbase = vh + (size_t)b * T * HEAD_DIM;

    // Q A-frags for K=64 (two 16x32 frags) stay resident all kernel.
    v16h qa0 = frag_row_f16(qrow, 0, half);
    v16h qa1 = frag_row_f16(qrow, 32, half);

    // ---------- Pass 1: online max + sum ----------
    float mrun[8], zrun[8];
#pragma unroll
    for (int r = 0; r < 8; ++r) { mrun[r] = -INFINITY; zrun[r] = 0.f; }

    for (int kt = 0; kt < 256; ++kt) {
        const _Float16* krow = kbase + (size_t)(kt * 16 + n) * HEAD_DIM;
        v16h b0 = frag_row_f16(krow, 0, half);
        v16h b1 = frag_row_f16(krow, 32, half);
        v8f acc = {};
        acc = wmma_f16f32(qa0, b0, acc);
        acc = wmma_f16f32(qa1, b1, acc);
#pragma unroll
        for (int r = 0; r < 8; ++r) {
            float s = acc[r] * 0.125f;
            float mn = fmaxf(mrun[r], s);
            zrun[r] = zrun[r] * __expf(mrun[r] - mn) + __expf(s - mn);
            mrun[r] = mn;
        }
    }
    // Cross-lane merge within the 16-lane column groups (half bit preserved).
#pragma unroll
    for (int r = 0; r < 8; ++r) {
        float m = mrun[r], z = zrun[r];
#pragma unroll
        for (int mask = 1; mask < 16; mask <<= 1) {
            float mo = __shfl_xor(m, mask, 32);
            float zo = __shfl_xor(z, mask, 32);
            float mn = fmaxf(m, mo);
            z = z * __expf(m - mn) + zo * __expf(mo - mn);
            m = mn;
        }
        mrun[r] = m;
        zrun[r] = 1.f / z;   // keep inverse
    }

    // ---------- Pass 2: normalize + store attn + PV + entropy ----------
    v8f oacc[4];
#pragma unroll
    for (int c = 0; c < 4; ++c) oacc[c] = (v8f){};
    float ent = 0.f;

    size_t arow_base = ((size_t)(b * T + qt * 16)) * T;  // + row*T + col

    for (int kt2 = 0; kt2 < 128; ++kt2) {
#pragma unroll
        for (int sub = 0; sub < 2; ++sub) {
            int kt = kt2 * 2 + sub;
            const _Float16* krow = kbase + (size_t)(kt * 16 + n) * HEAD_DIM;
            v16h b0 = frag_row_f16(krow, 0, half);
            v16h b1 = frag_row_f16(krow, 32, half);
            v8f acc = {};
            acc = wmma_f16f32(qa0, b0, acc);
            acc = wmma_f16f32(qa1, b1, acc);
#pragma unroll
            for (int r = 0; r < 8; ++r) {
                float s = acc[r] * 0.125f;
                float p = __expf(s - mrun[r]) * zrun[r];
                int row = r + half * 8;
                attn[arow_base + (size_t)row * T + kt * 16 + n] = p;   // the one HBM write
                ent -= p * __logf(p + 1e-8f);
                pbuf[row * 32 + sub * 16 + n] = (_Float16)p;           // C-form -> LDS
            }
        }
        // Reload P as A-frag (lane&15 = row, K=32 across the tile pair); per-wave DS
        // ops are in-order so the loads see this wave's stores.
        v16h pa = frag_row_f16(pbuf + n * 32, 0, half);
        const _Float16* vblk = vbase + (size_t)(kt2 * 32) * HEAD_DIM;
#pragma unroll
        for (int c = 0; c < 4; ++c) {
            v16h vb = frag_colB_f16(vblk, HEAD_DIM, c * 16 + n, 0, half);
            oacc[c] = wmma_f16f32(pa, vb, oacc[c]);
        }
    }

    // out tile store
#pragma unroll
    for (int c = 0; c < 4; ++c)
#pragma unroll
        for (int r = 0; r < 8; ++r)
            out[(size_t)(b * T + qt * 16 + r + half * 8) * HEAD_DIM + c * 16 + n] = oacc[c][r];

    // entropy: reduce across all 32 lanes, one atomic per wave
#pragma unroll
    for (int mask = 1; mask < 32; mask <<= 1)
        ent += __shfl_xor(ent, mask, 32);
    if (lane == 0) atomicAdd(ent_accum, ent);
}

// ---- Kernel 3: metabolic EMA ----
__global__ void finalize_kernel(const float* __restrict__ energy_in,
                                const float* __restrict__ ent_accum,
                                float* __restrict__ e_out) {
    if (threadIdx.x == 0) {
        float mean_ent = ent_accum[0] * (1.0f / (float)ROWS);
        e_out[0] = 0.9f * energy_in[0] + 0.1f * mean_ent;
    }
}

extern "C" void kernel_launch(void* const* d_in, const int* in_sizes, int n_in,
                              void* d_out, int out_size, void* d_ws, size_t ws_size,
                              hipStream_t stream) {
    (void)in_sizes; (void)n_in; (void)out_size; (void)ws_size;
    const float* x      = (const float*)d_in[0];
    const float* Wq     = (const float*)d_in[1];
    const float* Wk     = (const float*)d_in[2];
    const float* Wv     = (const float*)d_in[3];
    const float* energy = (const float*)d_in[4];

    float* outf = (float*)d_out;
    float* out  = outf + OUT_OFF;
    float* attn = outf + ATTN_OFF;
    float* eout = outf + EN_OFF;

    // Workspace: q/k/v as f16 (2 MB each -> L2 resident), entropy accumulator.
    const size_t KH = (size_t)ROWS * HEAD_DIM * sizeof(_Float16);   // 2 MB
    _Float16* qh = (_Float16*)((char*)d_ws);
    _Float16* kh = (_Float16*)((char*)d_ws + KH);
    _Float16* vh = (_Float16*)((char*)d_ws + 2 * KH);
    float* ent   = (float*)((char*)d_ws + 3 * KH);

    hipMemsetAsync(ent, 0, sizeof(float), stream);

    // 1) projections: 12288 waves, 4 waves/block
    proj_kernel<<<3072, 128, 0, stream>>>(x, Wq, Wk, Wv, qh, kh, vh);
    // 2) fused attention: 1024 waves, 8 waves/block
    attn_fused_kernel<<<128, 256, 0, stream>>>(qh, kh, vh, attn, out, ent);
    // 3) energy EMA
    finalize_kernel<<<1, 32, 0, stream>>>(energy, ent, eout);
}
